// SSPTransform_23304492548630
// MI455X (gfx1250) — matrix operand order repaired and checked by reading, not compiled, not run on hardware
//
#include <hip/hip_runtime.h>
#include <hip/hip_bf16.h>

typedef __attribute__((ext_vector_type(16))) _Float16 v16h;
typedef __attribute__((ext_vector_type(8)))  _Float16 v8h;
typedef __attribute__((ext_vector_type(8)))  float    v8f;

#define DIM    513
#define NPAR   256
#define KDIM   512
#define NPAD   528          // 33 tiles * 16
#define NT     33
#define BATCHN 131072
#define MB     128          // batches per block
#define LDSROW 520          // padded K stride in halves (bank-conflict dodge)

// ---------------------------------------------------------------------------
// Kernel 1: build transposed twiddle matrix W^T[n][j], f16, in workspace.
//   j <  256 :  (2/513) * cos(2*pi*(j+1)*n/513)
//   j >= 256 : -(2/513) * sin(2*pi*(j-255)*n/513)
//   rows n in [513,528) are zero padding.
// ---------------------------------------------------------------------------
__global__ void ssp_twiddle_kernel(_Float16* __restrict__ wt) {
    int idx = blockIdx.x * blockDim.x + threadIdx.x;   // idx = n*512 + j
    if (idx >= NPAD * KDIM) return;
    int n = idx >> 9;
    int j = idx & 511;
    float w = 0.0f;
    if (n < DIM) {
        int k = (j < NPAR) ? (j + 1) : (j - NPAR + 1);
        int t = (k * n) % DIM;                          // exact integer reduction
        float ang = 6.283185307179586f * (float)t / (float)DIM;
        float s, c;
        __sincosf(ang, &s, &c);
        const float scale = 2.0f / (float)DIM;
        w = (j < NPAR) ? (scale * c) : (-scale * s);
    }
    wt[idx] = (_Float16)w;
}

// ---------------------------------------------------------------------------
// Kernel 2: out[b,n] = 1/513 + A[b,:] . W[:,n] via v_wmma_f32_16x16x32_f16.
// Block = 128 batches, 8 waves; wave w owns M-tile w, sweeps 33 N-tiles.
// ---------------------------------------------------------------------------
__global__ void __launch_bounds__(256)
ssp_gemm_kernel(const float* __restrict__ x,
                const float* __restrict__ phis,
                const _Float16* __restrict__ wt,
                float* __restrict__ out) {
    __shared__ _Float16 Alds[MB * LDSROW];             // 133,120 bytes

    const int tid = threadIdx.x;
    const int b0  = blockIdx.x * MB;

    // ---- Build A tile: A[m][j] = j<256 ? cos(x_m*phi_j) : sin(x_m*phi_{j-256})
    {
        const int m    = tid >> 1;                     // 0..127
        const int half = tid & 1;                      // 0..1  (j range split)
        const float xv = x[b0 + m];
        _Float16* arow = Alds + m * LDSROW;
        #pragma unroll 4
        for (int q = 0; q < 128; ++q) {
            int j = half * 128 + q;
            float s, c;
            __sincosf(xv * phis[j], &s, &c);
            arow[j]        = (_Float16)c;
            arow[NPAR + j] = (_Float16)s;
        }
    }
    __syncthreads();

    const int wave   = tid >> 5;                       // M-tile within block
    const int lane   = tid & 31;
    const int laneN  = lane & 15;                      // N / M-row index
    const int hiHalf = lane >> 4;                      // K-half select

    // A fragment source: row = wave*16 + (lane&15); lane<16 gets K {0-7,16-23},
    // lane>=16 gets K {8-15,24-31}  -> base offset hiHalf*8, chunks at +0 / +16.
    const _Float16* a_base = Alds + (wave * 16 + laneN) * LDSROW + hiHalf * 8;

    for (int nt = 0; nt < NT; ++nt) {
        const int n = nt * 16 + laneN;
        // B fragment: lane holds K {0-7,16-23} (lanes 0-15) or
        // K {8-15,24-31} (lanes 16-31) of the step for column n.
        const _Float16* brow = wt + (size_t)n * KDIM + hiHalf * 8;

        v8f acc;
        #pragma unroll
        for (int i = 0; i < 8; ++i) acc[i] = 1.0f / (float)DIM;  // folded bias

        #pragma unroll 4
        for (int kk = 0; kk < 16; ++kk) {
            const int kb = kk * 32;
            v16h bfrag = __builtin_shufflevector(*(const v8h*)(brow + kb), *(const v8h*)(brow + kb + 16), 0,1,2,3,4,5,6,7,8,9,10,11,12,13,14,15);
            v8h  alo   = *(const v8h*)(a_base + kb);             // ds_read_b128
            v8h  ahi   = *(const v8h*)(a_base + kb + 16);        // ds_read_b128
            v16h afrag = __builtin_shufflevector(
                alo, ahi, 0,1,2,3,4,5,6,7,8,9,10,11,12,13,14,15);
            acc = __builtin_amdgcn_wmma_f32_16x16x32_f16(
                      false, afrag, false, bfrag, (short)0, acc, false, false);
        }

        if (n < DIM) {
            const int mrow = b0 + wave * 16 + hiHalf * 8;        // C layout:
            #pragma unroll                                       // VGPR v -> M=v(+8)
            for (int v = 0; v < 8; ++v)
                out[(size_t)(mrow + v) * DIM + n] = acc[v];
        }
    }
}

// ---------------------------------------------------------------------------
extern "C" void kernel_launch(void* const* d_in, const int* in_sizes, int n_in,
                              void* d_out, int out_size, void* d_ws, size_t ws_size,
                              hipStream_t stream) {
    (void)in_sizes; (void)n_in; (void)out_size; (void)ws_size;
    const float* xin  = (const float*)d_in[0];   // (131072, 1) f32
    const float* phis = (const float*)d_in[1];   // (256,) f32
    float*    outp = (float*)d_out;              // (131072, 513) f32
    _Float16* wt   = (_Float16*)d_ws;            // 528*512*2 = 540,672 B scratch

    ssp_twiddle_kernel<<<(NPAD * KDIM + 255) / 256, 256, 0, stream>>>(wt);
    ssp_gemm_kernel<<<BATCHN / MB, 256, 0, stream>>>(xin, phis, wt, outp);
}
